// SpanNERModel_28492813042215
// MI455X (gfx1250) — matrix operand order repaired and checked by reading, not compiled, not run on hardware
//
#include <hip/hip_runtime.h>
#include <hip/hip_bf16.h>

#define BB 32
#define SS 512
#define VV 30522
#define NSP 256
#define NLB 9
#define HH 768
#define NHEAD 12
#define DHEAD 64
#define NLAYER 12
#define FFD 3072

typedef __attribute__((ext_vector_type(16))) __bf16 v16bf;
typedef __attribute__((ext_vector_type(8)))  float  v8f;

union Frag { v16bf v; uint4 q[2]; };
static_assert(sizeof(Frag) == 32, "frag size");

__device__ inline v8f wmma_bf16(const Frag& a, const Frag& b, v8f c) {
  // D = A(16x32 bf16) * B(32x16 bf16) + C(16x16 f32)
  return __builtin_amdgcn_wmma_f32_16x16x32_bf16(
      /*neg_a=*/false, a.v, /*neg_b=*/false, b.v,
      /*c_mod=*/(short)0, c, /*reuse_a=*/false, /*reuse_b=*/false);
}

__device__ inline __bf16 f2bf(float f) {
  union { float f; unsigned u; } cv; cv.f = f;
  unsigned u = cv.u;
  u += 0x7FFFu + ((u >> 16) & 1u);   // round-to-nearest-even
  union { unsigned short s; __bf16 b; } o; o.s = (unsigned short)(u >> 16);
  return o.b;
}

// ---------------------------------------------------------------------------
// Tensor Data Mover: async 2D bf16 tile load (global -> LDS), with hardware
// row padding (tile_dim0 = 32 elems = 16 DWORDs, pad 4 DWORDs -> LDS stride 40)
// ---------------------------------------------------------------------------
#if defined(__has_builtin)
#if __has_builtin(__builtin_amdgcn_tensor_load_to_lds) && \
    __has_builtin(__builtin_amdgcn_s_wait_tensorcnt)
#define HAVE_TDM 1
#endif
#endif

#ifdef HAVE_TDM
typedef __attribute__((ext_vector_type(4))) unsigned tdm_v4u;
typedef __attribute__((ext_vector_type(8))) int      tdm_v8i;
typedef __attribute__((ext_vector_type(4))) int      tdm_v4i;

__device__ inline void tdm_load_tile(unsigned lds_off, const __bf16* gptr,
                                     unsigned tile_h,            // rows
                                     unsigned long long stride,  // row stride, elems
                                     unsigned long long tw,      // tensor dim0, elems
                                     unsigned long long th) {    // tensor dim1, rows
  unsigned long long ga = (unsigned long long)(uintptr_t)gptr;
  // D# group0: count=1 | lds_addr | global_addr[56:0] | type=2
  tdm_v4u g0 = { 1u, lds_off, (unsigned)ga,
                 (unsigned)((ga >> 32) & 0x01FFFFFFull) | (2u << 30) };
  // D# group1: data_size=2B, pad_enable, pad_interval=16dw, pad_amount=4dw,
  //            tensor_dim0/1, tile_dim0=32, tile_dim1=tile_h, dim0_stride
  tdm_v8i g1 = {
      (int)((1u << 16) | (1u << 20) | (3u << 22) | (3u << 25)),
      (int)((unsigned)(tw & 0xFFFFull) << 16),
      (int)((unsigned)((tw >> 16) & 0xFFFFull) | ((unsigned)(th & 0xFFFFull) << 16)),
      (int)((unsigned)((th >> 16) & 0xFFFFull) | (32u << 16)),
      (int)(tile_h & 0xFFFFu),
      (int)(unsigned)(stride & 0xFFFFFFFFull),
      (int)(unsigned)((stride >> 32) & 0xFFFFull),
      0 };
  tdm_v4i z4 = { 0, 0, 0, 0 };
#if __clang_major__ >= 23
  tdm_v8i z8 = { 0, 0, 0, 0, 0, 0, 0, 0 };
  __builtin_amdgcn_tensor_load_to_lds(g0, g1, z4, z4, z8, 0);
#else
  __builtin_amdgcn_tensor_load_to_lds(g0, g1, z4, z4, 0);
#endif
}
#endif

// ---------------------------------------------------------------------------
// fp32 [K][N] -> bf16 [N][K]  (K-major so WMMA B-fragments are contiguous)
// ---------------------------------------------------------------------------
__global__ __launch_bounds__(256) void convT(const float* __restrict__ W,
                                             __bf16* __restrict__ Wt,
                                             int K, int N) {
  long idx = (long)blockIdx.x * 256 + threadIdx.x;
  long total = (long)K * N;
  if (idx >= total) return;
  int k = (int)(idx / N);
  int n = (int)(idx % N);
  Wt[(long)n * K + k] = f2bf(W[idx]);
}

// ---------------------------------------------------------------------------
// C[M,N] = act(A[M,K](bf16) @ Wt[N,K](bf16)^T + bias[N])
// block = 128 thr (4 waves), block tile 64x128, wave tile 32x64, K-step 32.
// Double-buffered LDS; tiles DMA'd by the Tensor Data Mover (wave 0 issues,
// s_wait_tensorcnt + barrier for visibility) overlapping the WMMA pipeline.
// ---------------------------------------------------------------------------
__global__ __launch_bounds__(128) void gemm_bf16(
    const __bf16* __restrict__ A, const __bf16* __restrict__ Bt,
    const float* __restrict__ bias, float* __restrict__ outF,
    __bf16* __restrict__ outB, int M, int N, int K, int act) {
  __shared__ alignas(16) __bf16 As[2][64 * 40];    // [m][k], stride 40 (pad 8)
  __shared__ alignas(16) __bf16 Bs[2][128 * 40];   // [n][k], stride 40

  const int tid  = threadIdx.x;
  const int lane = tid & 31;
  const int w    = tid >> 5;
  const int n0 = blockIdx.x * 128;
  const int m0 = blockIdx.y * 64;
  const int wm = (w >> 1) * 32;
  const int wn = (w & 1) * 64;
  const int lm  = lane & 15;
  const int kb  = (lane >> 4) * 8;    // A-frag K group (groups of 8)
  const int kb2 = (lane >> 4) * 16;   // unused; B-frags also use kb

  v8f zero = {};
  v8f acc[2][4];
  for (int i = 0; i < 2; i++)
    for (int j = 0; j < 4; j++) acc[i][j] = zero;

  auto issue = [&](int buf, int k0) {
#ifdef HAVE_TDM
    if (w == 0) {
      tdm_load_tile((unsigned)(uintptr_t)&As[buf][0], &A[(long)m0 * K + k0],
                    64u, (unsigned long long)K,
                    (unsigned long long)(K - k0), (unsigned long long)(M - m0));
      tdm_load_tile((unsigned)(uintptr_t)&Bs[buf][0], &Bt[(long)n0 * K + k0],
                    128u, (unsigned long long)K,
                    (unsigned long long)(K - k0), (unsigned long long)(N - n0));
    }
#else
    for (int s = tid; s < 256; s += 128) {   // A: 64 rows x 4 segs of 8 bf16
      int row = s >> 2, seg = (s & 3) * 8;
      *reinterpret_cast<uint4*>(&As[buf][row * 40 + seg]) =
          *reinterpret_cast<const uint4*>(&A[(long)(m0 + row) * K + k0 + seg]);
    }
    for (int s = tid; s < 512; s += 128) {   // B: 128 rows x 4 segs of 8 bf16
      int row = s >> 2, seg = (s & 3) * 8;
      *reinterpret_cast<uint4*>(&Bs[buf][row * 40 + seg]) =
          *reinterpret_cast<const uint4*>(&Bt[(long)(n0 + row) * K + k0 + seg]);
    }
#endif
  };

  issue(0, 0);
  int buf = 0;
  for (int k0 = 0; k0 < K; k0 += 32, buf ^= 1) {
#ifdef HAVE_TDM
    if (w == 0) __builtin_amdgcn_s_wait_tensorcnt(0);  // current buf DMA done
#endif
    __syncthreads();          // data visible; prev compute done before overwrite
    if (k0 + 32 < K) issue(buf ^ 1, k0 + 32);

    Frag a[2], b[4];
    for (int mi = 0; mi < 2; mi++) {
      int r = wm + mi * 16 + lm;
      a[mi].q[0] = *reinterpret_cast<const uint4*>(&As[buf][r * 40 + kb]);
      a[mi].q[1] = *reinterpret_cast<const uint4*>(&As[buf][r * 40 + kb + 16]);
    }
    for (int ni = 0; ni < 4; ni++) {
      int r = wn + ni * 16 + lm;
      b[ni].q[0] = *reinterpret_cast<const uint4*>(&Bs[buf][r * 40 + kb]);
      b[ni].q[1] = *reinterpret_cast<const uint4*>(&Bs[buf][r * 40 + kb + 16]);
    }
    for (int mi = 0; mi < 2; mi++)
      for (int ni = 0; ni < 4; ni++)
        acc[mi][ni] = wmma_bf16(a[mi], b[ni], acc[mi][ni]);
  }

  const int rb = (lane >> 4) * 8;   // C/D layout: M = r + 8*(lane>=16)
  for (int mi = 0; mi < 2; mi++)
    for (int ni = 0; ni < 4; ni++)
      for (int r = 0; r < 8; r++) {
        int row = m0 + wm + mi * 16 + rb + r;
        int col = n0 + wn + ni * 16 + lm;
        float v = acc[mi][ni][r] + bias[col];
        if (act == 1) v = 0.5f * v * (1.0f + erff(v * 0.70710678118f));
        long o = (long)row * N + col;
        if (outF) outF[o] = v;
        if (outB) outB[o] = f2bf(v);
      }
}

// ---------------------------------------------------------------------------
// attention: block = (batch b, head h, 64-query tile). 128 thr = 4 waves.
// dynamic LDS: Sc 64x528 f32 | Qs 64x72 bf16 | Ks 64x72 | Vt 64x72 | bias 512 f32
// ---------------------------------------------------------------------------
#define SC_STRIDE 528
#define QS_OFF  135168
#define KS_OFF  144384
#define VT_OFF  153600
#define BIAS_OFF 162816
#define ATTN_SMEM (162816 + 2048)

__global__ __launch_bounds__(128) void attn_kernel(
    const __bf16* __restrict__ Qg, const __bf16* __restrict__ Kg,
    const __bf16* __restrict__ Vg, const int* __restrict__ mask,
    __bf16* __restrict__ ctx) {
  extern __shared__ char smem[];
  float*  Sc    = (float*)smem;
  __bf16* Qs    = (__bf16*)(smem + QS_OFF);
  __bf16* Ks    = (__bf16*)(smem + KS_OFF);
  __bf16* Vt    = (__bf16*)(smem + VT_OFF);
  float*  biasL = (float*)(smem + BIAS_OFF);

  const int tid = threadIdx.x, lane = tid & 31, w = tid >> 5;
  const int q0 = blockIdx.x * 64;
  const int hh = blockIdx.y;
  const int b  = blockIdx.z;
  const int lm  = lane & 15;
  const int kb  = (lane >> 4) * 8;
  const int kb2 = (lane >> 4) * 16;
  const long base = ((long)b * SS) * HH + (long)hh * DHEAD;

  for (int s = tid; s < 512; s += 128) {   // Q tile: 64 rows x 64 dh
    int row = s >> 3, seg = (s & 7) * 8;
    *reinterpret_cast<uint4*>(&Qs[row * 72 + seg]) =
        *reinterpret_cast<const uint4*>(&Qg[base + (long)(q0 + row) * HH + seg]);
  }
  for (int j = tid; j < SS; j += 128)
    biasL[j] = (1.0f - (float)mask[b * SS + j]) * -1.0e9f;
  __syncthreads();

  // ---- pass 1: S = Q K^T * scale + bias ----
  for (int kt = 0; kt < 8; kt++) {
    for (int s = tid; s < 512; s += 128) {
      int row = s >> 3, seg = (s & 7) * 8;
      *reinterpret_cast<uint4*>(&Ks[row * 72 + seg]) =
          *reinterpret_cast<const uint4*>(&Kg[base + (long)(kt * 64 + row) * HH + seg]);
    }
    __syncthreads();
    Frag aq[2];
    for (int ks = 0; ks < 2; ks++) {
      int r = w * 16 + lm;
      aq[ks].q[0] = *reinterpret_cast<const uint4*>(&Qs[r * 72 + ks * 32 + kb]);
      aq[ks].q[1] = *reinterpret_cast<const uint4*>(&Qs[r * 72 + ks * 32 + kb + 16]);
    }
    for (int ni = 0; ni < 4; ni++) {
      v8f acc = {};
      for (int ks = 0; ks < 2; ks++) {
        Frag bk;
        int r = ni * 16 + lm;
        bk.q[0] = *reinterpret_cast<const uint4*>(&Ks[r * 72 + ks * 32 + kb]);
        bk.q[1] = *reinterpret_cast<const uint4*>(&Ks[r * 72 + ks * 32 + kb + 16]);
        acc = wmma_bf16(aq[ks], bk, acc);
      }
      int rb = (lane >> 4) * 8;
      for (int r = 0; r < 8; r++) {
        int qrow = w * 16 + rb + r;
        int col  = kt * 64 + ni * 16 + lm;
        Sc[qrow * SC_STRIDE + col] = acc[r] * 0.125f + biasL[col];
      }
    }
    __syncthreads();
  }

  // ---- pass 2: row softmax, re-quantize P to bf16 in place ----
  if (tid < 64) {
    float* row = &Sc[tid * SC_STRIDE];
    float m = -3.0e38f;
    for (int j = 0; j < SS; j++) m = fmaxf(m, row[j]);
    float sum = 0.0f;
    for (int j = 0; j < SS; j++) { float e = __expf(row[j] - m); row[j] = e; sum += e; }
    float inv = 1.0f / sum;
    __bf16* prow = reinterpret_cast<__bf16*>(row);   // bf16 j clobbers f32 j/2 (<= j): safe
    for (int j = 0; j < SS; j++) prow[j] = f2bf(row[j] * inv);
  }
  __syncthreads();

  // ---- pass 3: ctx = P @ V ----
  v8f zero = {};
  v8f acc4[4];
  for (int i = 0; i < 4; i++) acc4[i] = zero;
  for (int vt = 0; vt < 8; vt++) {
    for (int s = tid; s < 4096; s += 128) {   // transpose V tile: [dh][key]
      int ki = s & 63, d = s >> 6;
      Vt[d * 72 + ki] = Vg[base + (long)(vt * 64 + ki) * HH + d];
    }
    __syncthreads();
    const unsigned short* P = reinterpret_cast<const unsigned short*>(Sc);
    for (int ks = 0; ks < 2; ks++) {
      Frag ap;
      int prow = (w * 16 + lm) * (SC_STRIDE * 2);
      int kpos = vt * 64 + ks * 32 + kb;
      ap.q[0] = *reinterpret_cast<const uint4*>(&P[prow + kpos]);
      ap.q[1] = *reinterpret_cast<const uint4*>(&P[prow + kpos + 16]);
      for (int ni = 0; ni < 4; ni++) {
        Frag bv;
        int r = ni * 16 + lm;
        bv.q[0] = *reinterpret_cast<const uint4*>(&Vt[r * 72 + ks * 32 + kb]);
        bv.q[1] = *reinterpret_cast<const uint4*>(&Vt[r * 72 + ks * 32 + kb + 16]);
        acc4[ni] = wmma_bf16(ap, bv, acc4[ni]);
      }
    }
    __syncthreads();
  }
  int rb = (lane >> 4) * 8;
  for (int ni = 0; ni < 4; ni++)
    for (int r = 0; r < 8; r++) {
      int qrow = q0 + w * 16 + rb + r;
      int col  = ni * 16 + lm;
      ctx[base + (long)qrow * HH + col] = f2bf(acc4[ni][r]);
    }
}

// ---------------------------------------------------------------------------
// embedding gather + LayerNorm  (one block per token, 256 thr, 3 elems/thr)
// ---------------------------------------------------------------------------
__global__ __launch_bounds__(256) void embed_ln(
    const int* __restrict__ ids, const float* __restrict__ tok,
    const float* __restrict__ pos, const float* __restrict__ typ,
    const float* __restrict__ g, const float* __restrict__ bb,
    float* __restrict__ h, __bf16* __restrict__ xb) {
  __shared__ float red[256];
  const int t = blockIdx.x;
  const int sidx = t % SS;
  const int id = ids[t];
  const int tid = threadIdx.x;
  float v[3];
  for (int i = 0; i < 3; i++) {
    int j = tid + i * 256;
    v[i] = tok[(long)id * HH + j] + pos[(long)sidx * HH + j] + typ[j];
  }
  red[tid] = v[0] + v[1] + v[2];
  __syncthreads();
  for (int off = 128; off > 0; off >>= 1) { if (tid < off) red[tid] += red[tid + off]; __syncthreads(); }
  float mean = red[0] * (1.0f / 768.0f);
  __syncthreads();
  float q = 0.0f;
  for (int i = 0; i < 3; i++) { float d = v[i] - mean; q += d * d; }
  red[tid] = q;
  __syncthreads();
  for (int off = 128; off > 0; off >>= 1) { if (tid < off) red[tid] += red[tid + off]; __syncthreads(); }
  float rstd = rsqrtf(red[0] * (1.0f / 768.0f) + 1e-12f);
  for (int i = 0; i < 3; i++) {
    int j = tid + i * 256;
    float y = (v[i] - mean) * rstd * g[j] + bb[j];
    h[(long)t * HH + j] = y;
    xb[(long)t * HH + j] = f2bf(y);
  }
}

// ---------------------------------------------------------------------------
// residual + LayerNorm:  h = LN(res + x), xb = bf16(h).  (res may alias hout)
// ---------------------------------------------------------------------------
__global__ __launch_bounds__(256) void res_ln(
    const float* res, const float* __restrict__ x,
    const float* __restrict__ g, const float* __restrict__ bb,
    float* hout, __bf16* __restrict__ xb) {
  __shared__ float red[256];
  const int t = blockIdx.x;
  const int tid = threadIdx.x;
  float v[3];
  for (int i = 0; i < 3; i++) {
    long j = (long)t * HH + tid + i * 256;
    v[i] = res[j] + x[j];
  }
  red[tid] = v[0] + v[1] + v[2];
  __syncthreads();
  for (int off = 128; off > 0; off >>= 1) { if (tid < off) red[tid] += red[tid + off]; __syncthreads(); }
  float mean = red[0] * (1.0f / 768.0f);
  __syncthreads();
  float q = 0.0f;
  for (int i = 0; i < 3; i++) { float d = v[i] - mean; q += d * d; }
  red[tid] = q;
  __syncthreads();
  for (int off = 128; off > 0; off >>= 1) { if (tid < off) red[tid] += red[tid + off]; __syncthreads(); }
  float rstd = rsqrtf(red[0] * (1.0f / 768.0f) + 1e-12f);
  for (int i = 0; i < 3; i++) {
    int j = tid + i * 256;
    float y = (v[i] - mean) * rstd * g[j] + bb[j];
    hout[(long)t * HH + j] = y;
    xb[(long)t * HH + j] = f2bf(y);
  }
}

// ---------------------------------------------------------------------------
// span gather + classifier: logits[b,n,:] = [h[s]; h[e]] @ W + bias
// ---------------------------------------------------------------------------
__global__ __launch_bounds__(128) void span_cls(
    const float* __restrict__ h, const int* __restrict__ spans,
    const float* __restrict__ W, const float* __restrict__ bias,
    float* __restrict__ out) {
  __shared__ float red[128];
  const int blk = blockIdx.x;
  const int b = blk / NSP;
  const int sIdx = spans[blk * 2 + 0];
  const int eIdx = spans[blk * 2 + 1];
  const float* hs = &h[((long)b * SS + sIdx) * HH];
  const float* he = &h[((long)b * SS + eIdx) * HH];
  const int tid = threadIdx.x;
  float acc[NLB];
  for (int c = 0; c < NLB; c++) acc[c] = 0.0f;
  for (int j = tid; j < 2 * HH; j += 128) {
    float x = (j < HH) ? hs[j] : he[j - HH];
    const float* wr = &W[(long)j * NLB];
    for (int c = 0; c < NLB; c++) acc[c] += x * wr[c];
  }
  for (int c = 0; c < NLB; c++) {
    red[tid] = acc[c];
    __syncthreads();
    for (int off = 64; off > 0; off >>= 1) { if (tid < off) red[tid] += red[tid + off]; __syncthreads(); }
    if (tid == 0) out[(long)blk * NLB + c] = red[0] + bias[c];
    __syncthreads();
  }
}

// ---------------------------------------------------------------------------
extern "C" void kernel_launch(void* const* d_in, const int* in_sizes, int n_in,
                              void* d_out, int out_size, void* d_ws, size_t ws_size,
                              hipStream_t stream) {
  const int*   ids  = (const int*)d_in[0];
  const int*   mask = (const int*)d_in[1];
  const int*   spans= (const int*)d_in[2];
  const float* tok  = (const float*)d_in[3];
  const float* pos  = (const float*)d_in[4];
  const float* typ  = (const float*)d_in[5];
  const float* eg   = (const float*)d_in[6];
  const float* ebi  = (const float*)d_in[7];
  const float* Wq   = (const float*)d_in[8];  const float* bq = (const float*)d_in[9];
  const float* Wk   = (const float*)d_in[10]; const float* bk = (const float*)d_in[11];
  const float* Wv   = (const float*)d_in[12]; const float* bv = (const float*)d_in[13];
  const float* Wo   = (const float*)d_in[14]; const float* bo = (const float*)d_in[15];
  const float* g1   = (const float*)d_in[16]; const float* be1= (const float*)d_in[17];
  const float* W1   = (const float*)d_in[18]; const float* b1 = (const float*)d_in[19];
  const float* W2   = (const float*)d_in[20]; const float* b2 = (const float*)d_in[21];
  const float* g2   = (const float*)d_in[22]; const float* be2= (const float*)d_in[23];
  const float* cW   = (const float*)d_in[24]; const float* cb = (const float*)d_in[25];
  float* out = (float*)d_out;

  const int M = BB * SS;                 // 16384 token rows
  char* ws = (char*)d_ws;
  size_t off = 0;
  auto carve = [&](size_t bytes) -> char* {
    char* p = ws + off;
    off += (bytes + 255) & ~(size_t)255;
    return p;
  };
  float*  h    = (float*) carve((size_t)M * HH * 4);
  __bf16* xb   = (__bf16*)carve((size_t)M * HH * 2);
  __bf16* qb   = (__bf16*)carve((size_t)M * HH * 2);
  __bf16* kbuf = (__bf16*)carve((size_t)M * HH * 2);
  __bf16* vbuf = (__bf16*)carve((size_t)M * HH * 2);
  __bf16* ctxb = (__bf16*)carve((size_t)M * HH * 2);
  float*  tmp  = (float*) carve((size_t)M * HH * 4);
  __bf16* ffb  = (__bf16*)carve((size_t)M * FFD * 2);
  __bf16* wst  = (__bf16*)carve((size_t)HH * FFD * 2);   // weight staging (bf16, transposed)

  embed_ln<<<M, 256, 0, stream>>>(ids, tok, pos, typ, eg, ebi, h, xb);

  const int cgH  = (HH * HH  + 255) / 256;
  const int cgHF = (HH * FFD + 255) / 256;
  const dim3 gH (HH  / 128, M / 64);    // 64x128 block tiles
  const dim3 gFF(FFD / 128, M / 64);

  for (int l = 0; l < NLAYER; l++) {
    const size_t wo = (size_t)l * HH * HH;
    const size_t fo = (size_t)l * HH * FFD;

    // Q, K, V projections (bf16 out)
    convT<<<cgH, 256, 0, stream>>>(Wq + wo, wst, HH, HH);
    gemm_bf16<<<gH, 128, 0, stream>>>(xb, wst, bq + l * HH, nullptr, qb, M, HH, HH, 0);
    convT<<<cgH, 256, 0, stream>>>(Wk + wo, wst, HH, HH);
    gemm_bf16<<<gH, 128, 0, stream>>>(xb, wst, bk + l * HH, nullptr, kbuf, M, HH, HH, 0);
    convT<<<cgH, 256, 0, stream>>>(Wv + wo, wst, HH, HH);
    gemm_bf16<<<gH, 128, 0, stream>>>(xb, wst, bv + l * HH, nullptr, vbuf, M, HH, HH, 0);

    // attention
    attn_kernel<<<dim3(SS / 64, NHEAD, BB), 128, ATTN_SMEM, stream>>>(
        qb, kbuf, vbuf, mask, ctxb);

    // output projection + residual LN1
    convT<<<cgH, 256, 0, stream>>>(Wo + wo, wst, HH, HH);
    gemm_bf16<<<gH, 128, 0, stream>>>(ctxb, wst, bo + l * HH, tmp, nullptr, M, HH, HH, 0);
    res_ln<<<M, 256, 0, stream>>>(h, tmp, g1 + l * HH, be1 + l * HH, h, xb);

    // FFN: gelu(x @ W1 + b1) @ W2 + b2, residual LN2
    convT<<<cgHF, 256, 0, stream>>>(W1 + fo, wst, HH, FFD);
    gemm_bf16<<<gFF, 128, 0, stream>>>(xb, wst, b1 + (size_t)l * FFD, nullptr, ffb, M, FFD, HH, 1);
    convT<<<cgHF, 256, 0, stream>>>(W2 + fo, wst, FFD, HH);
    gemm_bf16<<<gH, 128, 0, stream>>>(ffb, wst, b2 + l * HH, tmp, nullptr, M, HH, FFD, 0);
    res_ln<<<M, 256, 0, stream>>>(h, tmp, g2 + l * HH, be2 + l * HH, h, xb);
  }

  span_cls<<<BB * NSP, 128, 0, stream>>>(h, spans, cW, cb, out);
}
